// KANLinear_Haar_88802743812925
// MI455X (gfx1250) — matrix operand order repaired and checked by reading, not compiled, not run on hardware
//
#include <hip/hip_runtime.h>

// KAN-Haar layer for MI455X (gfx1250, wave32, WMMA).
// Fused GEMM: fold relu(x) in as a 32nd per-feature channel so
//   [B x (IN*32)] x [(IN*32) x OUT], one v_wmma_f32_16x16x32_bf16 per feature.
// Haar bases are in {-1,0,+1} -> exact in bf16; only weights lose mantissa.
// Mainloop is software-pipelined: register-double-buffered B fragments so the
// loads for feature f+1 are in flight while feature f's 8 WMMAs execute.
// Workspace: [0,1KB) min (ord-uint), [1KB,2KB) max, [2KB, 2KB+4MB) packed bf16 W.

#define NB      31
#define BATCH   16384
#define INF_    256
#define OUTF    256
#define MTILE   64
#define XSTR    258   // LDS row stride (elements): 258 -> conflict-free A reads
#define WSLICE  (16 * 32 * 16)  // ushorts per feature slice of packed W (16 KB)

typedef __attribute__((ext_vector_type(16))) __bf16 v16bf;
typedef __attribute__((ext_vector_type(8)))  float  v8f;

__device__ __forceinline__ unsigned enc_ord(float f) {
  unsigned u = __float_as_uint(f);
  return (u & 0x80000000u) ? ~u : (u | 0x80000000u);
}
__device__ __forceinline__ float dec_ord(unsigned u) {
  u = (u & 0x80000000u) ? (u & 0x7FFFFFFFu) : ~u;
  return __uint_as_float(u);
}

// Haar basis k (compile-time): k = (2^lvl - 1) + shift.
// cf[lvl] = floor(xn * 2^lvl), sg[lvl] = (frac < 0.5 ? +1 : -1):
// basis_k = (cf[lvl] == shift) ? sg[lvl] : 0.
template <int K>
__device__ __forceinline__ __bf16 haar(const float* cf, const __bf16* sg) {
  constexpr int lvl   = (K < 1) ? 0 : (K < 3) ? 1 : (K < 7) ? 2 : (K < 15) ? 3 : 4;
  constexpr int shift = K - ((1 << lvl) - 1);
  return (cf[lvl] == (float)shift) ? sg[lvl] : (__bf16)0.0f;
}

// ---------------- kernel 0: init ordered-uint min/max ----------------
__global__ void kan_init_kernel(unsigned* __restrict__ minU, unsigned* __restrict__ maxU) {
  int t = threadIdx.x;
  minU[t] = 0xFFFFFFFFu;
  maxU[t] = 0u;
}

// ---------------- kernel 1: per-column min/max over the batch ----------------
__global__ __launch_bounds__(256) void kan_minmax_kernel(const float* __restrict__ x,
                                                         unsigned* __restrict__ minU,
                                                         unsigned* __restrict__ maxU) {
  int t  = threadIdx.x;              // column
  int r0 = blockIdx.x * 256;         // row chunk
  unsigned em = 0xFFFFFFFFu, eM = 0u;
#pragma unroll 4
  for (int r = 0; r < 256; ++r) {
    unsigned u = enc_ord(x[(size_t)(r0 + r) * INF_ + t]);  // coalesced
    em = min(em, u);
    eM = max(eM, u);
  }
  atomicMin(&minU[t], em);
  atomicMax(&maxU[t], eM);
}

// ---------------- kernel 2: pack weights into per-lane B-fragment order ----------------
// Element index = ((i*16 + ntile)*32 + lane)*16 + h ; lane<16 -> K={0..7,16..23}, lane>=16 -> K={8..15,24..31}.
// K==31 carries base_weight (relu channel); else spline_weight * spline_scaler.
__global__ __launch_bounds__(256) void kan_pack_kernel(const float* __restrict__ bw,
                                                       const float* __restrict__ sw,
                                                       const float* __restrict__ sc,
                                                       unsigned short* __restrict__ Wp) {
  unsigned tid = blockIdx.x * 256u + threadIdx.x;   // < 2^21
  int h    = tid & 15;
  int lane = (tid >> 4) & 31;
  int nt   = (tid >> 9) & 15;
  int i    = tid >> 13;
  int o    = nt * 16 + (lane & 15);
  int k    = ((lane & 16) ? 8 : 0) + ((h & 8) ? 16 : 0) + (h & 7);
  size_t oi = (size_t)o * INF_ + i;
  float v = (k < NB) ? sw[oi * NB + k] * sc[oi] : bw[oi];
  __bf16 b = (__bf16)v;
  Wp[tid] = __builtin_bit_cast(unsigned short, b);
}

// ---------------- helpers for the fused WMMA mainloop ----------------
__device__ __forceinline__ void loadB(v16bf dst[8], const unsigned short* baseLane, int f) {
  const unsigned short* p = baseLane + (size_t)f * WSLICE;
#pragma unroll
  for (int n = 0; n < 8; ++n)
    dst[n] = *(const v16bf*)(p + (size_t)n * 32 * 16);   // 32 B/lane contiguous
}

__device__ __forceinline__ void wmma8(v8f acc[8], const v16bf& a, const v16bf b[8]) {
#pragma unroll
  for (int n = 0; n < 8; ++n)
    acc[n] = __builtin_amdgcn_wmma_f32_16x16x32_bf16(
        /*neg_a=*/false, a, /*neg_b=*/false, b[n],
        /*c_mod=*/(short)0, acc[n], /*reuse_a=*/false, /*reuse_b=*/false);
}

__device__ __forceinline__ v16bf makeA(const __bf16* __restrict__ xnRow,
                                       const float2* __restrict__ mr,
                                       int f, bool hi) {
  float  xn  = (float)xnRow[f];
  float2 m   = mr[f];
  __bf16 rxb = (__bf16)fmaxf(fmaf(xn, m.y, m.x), 0.0f);   // relu(x) channel

  float  cf[5];
  __bf16 sg[5];
  const __bf16 onep = (__bf16)1.0f;
  const __bf16 onen = (__bf16)-1.0f;
#pragma unroll
  for (int l = 0; l < 5; ++l) {
    float tv = xn * (float)(1 << l);
    float c  = floorf(tv);
    cf[l] = c;
    sg[l] = (tv - c < 0.5f) ? onep : onen;
  }

  // A-fragment (16-bit A 16x32 layout): lane<16 -> K = {0..7,16..23},
  // lane>=16 -> K = {8..15,24..31}; K==31 is the relu channel.
  v16bf a;
#define HAAR_PAIR(H, KA, KB)                                      \
  { __bf16 vA = haar<KA>(cf, sg); __bf16 vB = haar<KB>(cf, sg);   \
    a[H] = hi ? vB : vA; }
  HAAR_PAIR(0, 0, 8)    HAAR_PAIR(1, 1, 9)    HAAR_PAIR(2, 2, 10)  HAAR_PAIR(3, 3, 11)
  HAAR_PAIR(4, 4, 12)   HAAR_PAIR(5, 5, 13)   HAAR_PAIR(6, 6, 14)  HAAR_PAIR(7, 7, 15)
  HAAR_PAIR(8, 16, 24)  HAAR_PAIR(9, 17, 25)  HAAR_PAIR(10, 18, 26) HAAR_PAIR(11, 19, 27)
  HAAR_PAIR(12, 20, 28) HAAR_PAIR(13, 21, 29) HAAR_PAIR(14, 22, 30)
  { __bf16 vA = haar<23>(cf, sg); a[15] = hi ? rxb : vA; }
#undef HAAR_PAIR
  return a;
}

// ---------------- kernel 3: fused WMMA GEMM (software-pipelined) ----------------
__global__ __launch_bounds__(256) void kan_wmma_kernel(const float* __restrict__ x,
                                                       const unsigned* __restrict__ minU,
                                                       const unsigned* __restrict__ maxU,
                                                       const unsigned short* __restrict__ Wp,
                                                       float* __restrict__ out) {
  __shared__ __bf16 xnLds[MTILE * XSTR];   // ~33 KB normalized x tile (bf16)
  __shared__ float2 mrLds[INF_];           // per-feature {min, range}

  const int t  = threadIdx.x;
  const int m0 = blockIdx.x * MTILE;

  // Normalize this block's 64-row x tile into LDS.
  {
    float mn  = dec_ord(minU[t]);
    float mx  = dec_ord(maxU[t]);
    float rng = mx - mn + 1e-8f;
    float inv = 1.0f / rng;
    mrLds[t] = make_float2(mn, rng);
#pragma unroll 4
    for (int r = 0; r < MTILE; ++r) {
      float f = x[(size_t)(m0 + r) * INF_ + t];      // coalesced
      xnLds[r * XSTR + t] = (__bf16)((f - mn) * inv);
    }
  }
  __syncthreads();

  const int  wave   = t >> 5;
  const int  lane   = t & 31;
  const bool hi     = lane >= 16;
  const int  msub   = wave & 3;          // 4 M-subtiles of 16 rows
  const int  nBase  = (wave >> 2) * 8;   // 8 N-tiles per wave (tile units)
  const __bf16* xnRow = xnLds + (msub * 16 + (lane & 15)) * XSTR;
  const unsigned short* bLane = Wp + ((size_t)nBase * 32 + lane) * 16;

  v8f acc[8];
#pragma unroll
  for (int n = 0; n < 8; ++n)
#pragma unroll
    for (int j = 0; j < 8; ++j) acc[n][j] = 0.0f;

  // Software pipeline: b0 holds even features, b1 holds odd features.
  v16bf b0[8], b1[8];
  loadB(b0, bLane, 0);

  int i = 0;
#pragma unroll 1
  for (; i + 2 < INF_; i += 2) {
    loadB(b1, bLane, i + 1);                       // in flight over wmma8(b0)
    __builtin_prefetch((const void*)(bLane + (size_t)(i + 6) * WSLICE), 0, 1);
    v16bf a0 = makeA(xnRow, mrLds, i, hi);
    wmma8(acc, a0, b0);
    loadB(b0, bLane, i + 2);                       // in flight over wmma8(b1)
    v16bf a1 = makeA(xnRow, mrLds, i + 1, hi);
    wmma8(acc, a1, b1);
  }
  // Epilogue: features 254, 255 (no loads past the packed-W buffer).
  {
    loadB(b1, bLane, i + 1);
    v16bf a0 = makeA(xnRow, mrLds, i, hi);
    wmma8(acc, a0, b0);
    v16bf a1 = makeA(xnRow, mrLds, i + 1, hi);
    wmma8(acc, a1, b1);
  }

  // D layout: lanes 0..15 -> N=lane, M=j ; lanes 16..31 -> N=lane-16, M=8+j.
  const int rowBase = m0 + msub * 16 + (hi ? 8 : 0);
  const int colBase = nBase * 16 + (lane & 15);
#pragma unroll
  for (int n = 0; n < 8; ++n)
#pragma unroll
    for (int j = 0; j < 8; ++j)
      __builtin_nontemporal_store(acc[n][j],
          out + (size_t)(rowBase + j) * OUTF + colBase + n * 16);
}

extern "C" void kernel_launch(void* const* d_in, const int* in_sizes, int n_in,
                              void* d_out, int out_size, void* d_ws, size_t ws_size,
                              hipStream_t stream) {
  (void)in_sizes; (void)n_in; (void)out_size; (void)ws_size;
  const float* x  = (const float*)d_in[0];
  const float* bw = (const float*)d_in[1];
  const float* sw = (const float*)d_in[2];
  const float* sc = (const float*)d_in[3];
  float* out = (float*)d_out;

  unsigned* minU = (unsigned*)d_ws;
  unsigned* maxU = minU + INF_;
  unsigned short* Wp = (unsigned short*)((char*)d_ws + 2048);  // 4 MB packed bf16 W

  kan_init_kernel<<<1, 256, 0, stream>>>(minU, maxU);
  kan_minmax_kernel<<<BATCH / 256, 256, 0, stream>>>(x, minU, maxU);
  kan_pack_kernel<<<(INF_ * 16 * 32 * 16) / 256, 256, 0, stream>>>(bw, sw, sc, Wp);
  kan_wmma_kernel<<<BATCH / MTILE, 256, 0, stream>>>(x, minU, maxU, Wp, out);
}
